// GNNReasoner_67577015435506
// MI455X (gfx1250) — matrix operand order repaired and checked by reading, not compiled, not run on hardware
//
#include <hip/hip_runtime.h>

// ---------------- problem constants (match reference) ----------------
#define NN      4096
#define EE      131072
#define D_IN    768
#define D_H     256
#define NHEAD   4
#define D_HEAD  64
#define D_OUT   128
#define NLAYER  2

typedef __bf16 bf16_t;
typedef bf16_t v16bf __attribute__((ext_vector_type(16)));
typedef float  v8f   __attribute__((ext_vector_type(8)));

__device__ __forceinline__ unsigned short bf16_rne(float f) {
    unsigned int u = __float_as_uint(f);
    unsigned int r = u + 0x7FFFu + ((u >> 16) & 1u);
    return (unsigned short)(r >> 16);
}

// ---------------- conversion kernels ----------------
__global__ void conv_f32_bf16(const float* __restrict__ in,
                              unsigned short* __restrict__ out, int n) {
    int i = blockIdx.x * 256 + threadIdx.x;
    if (i < n) out[i] = bf16_rne(in[i]);
}

// W[K,Nc] (row-major) -> Bt[Nc,K] bf16
__global__ void conv_transpose_bf16(const float* __restrict__ W,
                                    unsigned short* __restrict__ out, int K, int Nc) {
    int i = blockIdx.x * 256 + threadIdx.x;
    if (i >= K * Nc) return;
    int k = i / Nc, n = i % Nc;
    out[(size_t)n * K + k] = bf16_rne(W[i]);
}

// Wg_l[H,D_H,D_HEAD] -> Bt[(h*64+o), i] bf16  (fused per-head projection weight)
__global__ void conv_wg_bf16(const float* __restrict__ Wg_l,
                             unsigned short* __restrict__ out) {
    int idx = blockIdx.x * 256 + threadIdx.x;      // over 4*256*64 = 65536
    if (idx >= NHEAD * D_H * D_HEAD) return;
    int h = idx >> 14;          // / (256*64)
    int rem = idx & 16383;
    int i = rem >> 6;           // / 64
    int o = rem & 63;
    out[(size_t)(h * D_HEAD + o) * D_H + i] = bf16_rne(Wg_l[idx]);
}

// ---------------- WMMA GEMM: C[M,Nc] = act(A[M,K](bf16) * Bt[Nc,K](bf16) + bias) ----
// Each wave: one 16x64 output strip (4 N-subtiles), A fragment reused 4x per K-step.
__global__ __launch_bounds__(256) void wmma_gemm_bf16(
    const unsigned short* __restrict__ A,    // [M,K] bf16 row-major
    const unsigned short* __restrict__ Bt,   // [Nc,K] bf16 (B transposed)
    const float* __restrict__ bias,          // [Nc] or nullptr
    float* __restrict__ C,                   // [M,Nc] f32
    int M, int Nc, int K, int do_relu)
{
    const int lane = threadIdx.x & 31;
    const int wid  = threadIdx.x >> 5;
    const int ngrp = Nc >> 6;                          // groups of 64 cols
    const int wglobal = blockIdx.x * 8 + wid;
    const int total = (M >> 4) * ngrp;
    if (wglobal >= total) return;                      // wave-uniform
    const int tm = (wglobal / ngrp) << 4;
    const int tn = (wglobal % ngrp) << 6;

    const int half = lane >> 4;
    const int mn   = lane & 15;

    // A fragment layout: lane(m=mn), element e -> K = half*16 + e (contiguous)
    const unsigned short* arow = A + (size_t)(tm + mn) * K + half * 16;
    // B fragment layout: lane(n=mn), element e -> k = half*16 + e (contiguous)
    const unsigned short* bp0 = Bt + (size_t)(tn +  0 + mn) * K + half * 16;
    const unsigned short* bp1 = Bt + (size_t)(tn + 16 + mn) * K + half * 16;
    const unsigned short* bp2 = Bt + (size_t)(tn + 32 + mn) * K + half * 16;
    const unsigned short* bp3 = Bt + (size_t)(tn + 48 + mn) * K + half * 16;

    v8f acc0 = {0,0,0,0,0,0,0,0}, acc1 = acc0, acc2 = acc0, acc3 = acc0;

    union Frag { v16bf v; uint4 q[2]; };

    for (int k0 = 0; k0 < K; k0 += 32) {
        Frag fa;
        fa.q[0] = *(const uint4*)(arow + k0);
        fa.q[1] = *(const uint4*)(arow + k0 + 8);
        Frag fb;
        fb.q[0] = *(const uint4*)(bp0 + k0);
        fb.q[1] = *(const uint4*)(bp0 + k0 + 8);
        acc0 = __builtin_amdgcn_wmma_f32_16x16x32_bf16(false, fa.v, false, fb.v,
                                                       (short)0, acc0, false, false);
        fb.q[0] = *(const uint4*)(bp1 + k0);
        fb.q[1] = *(const uint4*)(bp1 + k0 + 8);
        acc1 = __builtin_amdgcn_wmma_f32_16x16x32_bf16(false, fa.v, false, fb.v,
                                                       (short)0, acc1, false, false);
        fb.q[0] = *(const uint4*)(bp2 + k0);
        fb.q[1] = *(const uint4*)(bp2 + k0 + 8);
        acc2 = __builtin_amdgcn_wmma_f32_16x16x32_bf16(false, fa.v, false, fb.v,
                                                       (short)0, acc2, false, false);
        fb.q[0] = *(const uint4*)(bp3 + k0);
        fb.q[1] = *(const uint4*)(bp3 + k0 + 8);
        acc3 = __builtin_amdgcn_wmma_f32_16x16x32_bf16(false, fa.v, false, fb.v,
                                                       (short)0, acc3, false, false);
    }

    float b0 = 0.f, b1 = 0.f, b2 = 0.f, b3 = 0.f;
    if (bias) {
        b0 = bias[tn +  0 + mn]; b1 = bias[tn + 16 + mn];
        b2 = bias[tn + 32 + mn]; b3 = bias[tn + 48 + mn];
    }
    float* crow = C + (size_t)tm * Nc + tn;
    #pragma unroll
    for (int r = 0; r < 8; ++r) {
        int m = r + half * 8;                 // C/D layout: VGPR r, half -> row
        float* cp = crow + (size_t)m * Nc + mn;
        float v0 = acc0[r] + b0, v1 = acc1[r] + b1;
        float v2 = acc2[r] + b2, v3 = acc3[r] + b3;
        if (do_relu) {
            v0 = fmaxf(v0, 0.f); v1 = fmaxf(v1, 0.f);
            v2 = fmaxf(v2, 0.f); v3 = fmaxf(v3, 0.f);
        }
        cp[0] = v0; cp[16] = v1; cp[32] = v2; cp[48] = v3;
    }
}

// ---------------- attention pieces ----------------
// s_src[h,n], s_dst[h,n] : 64-length dots of hh row segments with a-vectors
__global__ void ssd_kernel(const float* __restrict__ hh,      // [N, 256]
                           const float* __restrict__ a_l,     // [H, 128]
                           float* __restrict__ s_src, float* __restrict__ s_dst) {
    int idx = blockIdx.x * 256 + threadIdx.x;                 // h*N + n
    if (idx >= NHEAD * NN) return;
    int h = idx / NN, n = idx % NN;
    const float* hv = hh + (size_t)n * D_H + h * D_HEAD;
    const float* as = a_l + h * 2 * D_HEAD;
    const float* ad = as + D_HEAD;
    float ss = 0.f, sd = 0.f;
    #pragma unroll 8
    for (int o = 0; o < D_HEAD; ++o) { float v = hv[o]; ss += v * as[o]; sd += v * ad[o]; }
    s_src[idx] = ss; s_dst[idx] = sd;
}

// S_all[c] = sum_n hh[n, c]   (rank-1 softmax correction term)
__global__ void colsum_kernel(const float* __restrict__ hh, float* __restrict__ S_all) {
    __shared__ float red[256];
    int c = blockIdx.x;                                       // 0..255
    float s = 0.f;
    for (int r = threadIdx.x; r < NN; r += 256) s += hh[(size_t)r * D_H + c];
    red[threadIdx.x] = s; __syncthreads();
    for (int st = 128; st > 0; st >>= 1) {
        if (threadIdx.x < st) red[threadIdx.x] += red[threadIdx.x + st];
        __syncthreads();
    }
    if (threadIdx.x == 0) S_all[c] = red[0];
}

// pass 1: mx[h,row] = max(0, max_edges leaky_relu(...)); nonneg floats order as uints
__global__ void attn_max_kernel(const int* __restrict__ ei,
                                const float* __restrict__ s_src,
                                const float* __restrict__ s_dst,
                                unsigned int* __restrict__ mxbits) {
    int idx = blockIdx.x * 256 + threadIdx.x;
    if (idx >= EE * NHEAD) return;
    int h = idx & 3, e = idx >> 2;
    int row = ei[e], col = ei[EE + e];
    float v = s_src[h * NN + row] + s_dst[h * NN + col];
    float lv = v > 0.f ? v : 0.2f * v;
    if (lv > 0.f) atomicMax(&mxbits[h * NN + row], __float_as_uint(lv));
}

// pass 2: one block per edge; lane = (h, o). w = exp(e-mx) - exp(-mx)
__global__ void attn_acc_kernel(const int* __restrict__ ei,
                                const float* __restrict__ s_src,
                                const float* __restrict__ s_dst,
                                const float* __restrict__ mx,
                                const float* __restrict__ hh,
                                float* __restrict__ numer, float* __restrict__ Zpart) {
    int e = blockIdx.x;
    int h = threadIdx.x >> 6, o = threadIdx.x & 63;
    int row = ei[e], col = ei[EE + e];
    float v = s_src[h * NN + row] + s_dst[h * NN + col];
    float lv = v > 0.f ? v : 0.2f * v;
    float m  = mx[h * NN + row];
    float w  = __expf(lv - m) - __expf(-m);
    atomicAdd(&numer[((size_t)h * NN + row) * D_HEAD + o],
              w * hh[(size_t)col * D_H + h * D_HEAD + o]);
    if (o == 0) atomicAdd(&Zpart[h * NN + row], w);
}

// finalize: h_out[n, h*64+o] = relu( (numer + exp(-mx)*S_all) / (Zpart + N*exp(-mx)) )
__global__ void attn_final_kernel(const float* __restrict__ numer,
                                  const float* __restrict__ Zpart,
                                  const float* __restrict__ mx,
                                  const float* __restrict__ S_all,
                                  float* __restrict__ hout) {
    int idx = blockIdx.x * 256 + threadIdx.x;                 // n*256 + c
    if (idx >= NN * D_H) return;
    int n = idx >> 8, c = idx & 255;
    int h = c >> 6, o = c & 63;
    float m   = mx[h * NN + n];
    float emx = __expf(-m);
    float Z   = Zpart[h * NN + n] + (float)NN * emx;
    float num = numer[((size_t)h * NN + n) * D_HEAD + o] + emx * S_all[c];
    float r = num / Z;
    hout[idx] = r > 0.f ? r : 0.f;
}

// ---------------- host orchestration ----------------
static inline int cdiv(int a, int b) { return (a + b - 1) / b; }
static inline size_t alignup(size_t x) { return (x + 255) & ~(size_t)255; }

extern "C" void kernel_launch(void* const* d_in, const int* in_sizes, int n_in,
                              void* d_out, int out_size, void* d_ws, size_t ws_size,
                              hipStream_t stream) {
    const float* x    = (const float*)d_in[0];
    const int*   ei   = (const int*)  d_in[1];   // edge_index [2,E]
    const float* Win  = (const float*)d_in[2];   // [768,256]
    const float* binp = (const float*)d_in[3];   // [256]
    const float* Wg   = (const float*)d_in[4];   // [L,H,256,64]
    const float* av   = (const float*)d_in[5];   // [L,H,128]
    const float* Wout = (const float*)d_in[6];   // [256,128]
    const float* bout = (const float*)d_in[7];   // [128]
    float* out = (float*)d_out;                  // [N,128]

    // workspace carve-out (~22 MB)
    char* ws = (char*)d_ws;
    size_t off = 0;
    unsigned short* xbf   = (unsigned short*)(ws + off); off = alignup(off + (size_t)NN * D_IN * 2);
    unsigned short* WinT  = (unsigned short*)(ws + off); off = alignup(off + (size_t)D_H * D_IN * 2);
    unsigned short* WgT0  = (unsigned short*)(ws + off); off = alignup(off + (size_t)D_H * D_H * 2);
    unsigned short* WgT1  = (unsigned short*)(ws + off); off = alignup(off + (size_t)D_H * D_H * 2);
    unsigned short* WoutT = (unsigned short*)(ws + off); off = alignup(off + (size_t)D_OUT * D_H * 2);
    float*          hbuf  = (float*)(ws + off);          off = alignup(off + (size_t)NN * D_H * 4);
    unsigned short* hbf   = (unsigned short*)(ws + off); off = alignup(off + (size_t)NN * D_H * 2);
    float*          hh    = (float*)(ws + off);          off = alignup(off + (size_t)NN * D_H * 4);
    float*          ssrc  = (float*)(ws + off);          off = alignup(off + (size_t)NHEAD * NN * 4);
    float*          sdst  = (float*)(ws + off);          off = alignup(off + (size_t)NHEAD * NN * 4);
    float*          mx    = (float*)(ws + off);          off = alignup(off + (size_t)NHEAD * NN * 4);
    float*          Zp    = (float*)(ws + off);          off = alignup(off + (size_t)NHEAD * NN * 4);
    float*          Sall  = (float*)(ws + off);          off = alignup(off + (size_t)D_H * 4);
    float*          numer = (float*)(ws + off);          off = alignup(off + (size_t)NHEAD * NN * D_HEAD * 4);
    (void)ws_size; (void)n_in; (void)in_sizes; (void)out_size;

    // ---- weight / input conversions to bf16 (B kept transposed: [Nc,K]) ----
    conv_f32_bf16<<<cdiv(NN * D_IN, 256), 256, 0, stream>>>(x, xbf, NN * D_IN);
    conv_transpose_bf16<<<cdiv(D_IN * D_H, 256), 256, 0, stream>>>(Win, WinT, D_IN, D_H);
    conv_wg_bf16<<<cdiv(NHEAD * D_H * D_HEAD, 256), 256, 0, stream>>>(Wg, WgT0);
    conv_wg_bf16<<<cdiv(NHEAD * D_H * D_HEAD, 256), 256, 0, stream>>>(Wg + NHEAD * D_H * D_HEAD, WgT1);
    conv_transpose_bf16<<<cdiv(D_H * D_OUT, 256), 256, 0, stream>>>(Wout, WoutT, D_H, D_OUT);

    // ---- stage A: h = relu(x @ Win + bin) ----
    {
        int waves = (NN / 16) * (D_H / 64);
        wmma_gemm_bf16<<<cdiv(waves, 8), 256, 0, stream>>>(xbf, WinT, binp, hbuf,
                                                           NN, D_H, D_IN, 1);
    }

    // ---- GAT layers ----
    unsigned short* WgT[2] = { WgT0, WgT1 };
    for (int l = 0; l < NLAYER; ++l) {
        conv_f32_bf16<<<cdiv(NN * D_H, 256), 256, 0, stream>>>(hbuf, hbf, NN * D_H);
        int waves = (NN / 16) * (D_H / 64);
        wmma_gemm_bf16<<<cdiv(waves, 8), 256, 0, stream>>>(hbf, WgT[l], nullptr, hh,
                                                           NN, D_H, D_H, 0);
        ssd_kernel<<<cdiv(NHEAD * NN, 256), 256, 0, stream>>>(hh, av + l * NHEAD * 2 * D_HEAD,
                                                              ssrc, sdst);
        colsum_kernel<<<D_H, 256, 0, stream>>>(hh, Sall);
        hipMemsetAsync(mx,    0, (size_t)NHEAD * NN * 4, stream);
        hipMemsetAsync(Zp,    0, (size_t)NHEAD * NN * 4, stream);
        hipMemsetAsync(numer, 0, (size_t)NHEAD * NN * D_HEAD * 4, stream);
        attn_max_kernel<<<cdiv(EE * NHEAD, 256), 256, 0, stream>>>(ei, ssrc, sdst,
                                                                   (unsigned int*)mx);
        attn_acc_kernel<<<EE, 256, 0, stream>>>(ei, ssrc, sdst, mx, hh, numer, Zp);
        attn_final_kernel<<<cdiv(NN * D_H, 256), 256, 0, stream>>>(numer, Zp, mx, Sall, hbuf);
    }

    // ---- stage C: out = h @ Wout + bout ----
    conv_f32_bf16<<<cdiv(NN * D_H, 256), 256, 0, stream>>>(hbuf, hbf, NN * D_H);
    {
        int waves = (NN / 16) * (D_OUT / 64);
        wmma_gemm_bf16<<<cdiv(waves, 8), 256, 0, stream>>>(hbf, WoutT, bout, out,
                                                           NN, D_OUT, D_H, 0);
    }
}
